// Attention_org_29961691856973
// MI455X (gfx1250) — matrix operand order repaired and checked by reading, not compiled, not run on hardware
//
#include <hip/hip_runtime.h>
#include <hip/hip_bf16.h>
#include <math.h>

typedef __attribute__((ext_vector_type(16))) __bf16 v16bf;
typedef __attribute__((ext_vector_type(8)))  float  v8f;
typedef __attribute__((ext_vector_type(4)))  unsigned int uint4v;
typedef __attribute__((ext_vector_type(4)))  int     int4v;

#define TPB 256
#define GEMM_TPB 128

// Async global->LDS copy path (CDNA5 GLOBAL_LOAD_ASYNC_TO_LDS_B128, ASYNCcnt).
#if defined(__AMDGCN__) && __has_builtin(__builtin_amdgcn_global_load_async_to_lds_b128)
#define ASYNC_COPY 1
#else
#define ASYNC_COPY 0
#endif

__device__ __forceinline__ void copy16_g2l(const __bf16* g, __bf16* l) {
#if ASYNC_COPY
  __builtin_amdgcn_global_load_async_to_lds_b128(
      (__attribute__((address_space(1))) int4v*)(void*)g,
      (__attribute__((address_space(3))) int4v*)(void*)l, 0, 0);
#else
  *(uint4v*)l = *(const uint4v*)g;
#endif
}

__device__ __forceinline__ void wait_async_lds() {
#if ASYNC_COPY
#if __has_builtin(__builtin_amdgcn_s_wait_asynccnt)
  __builtin_amdgcn_s_wait_asynccnt(0);
#else
  asm volatile("s_wait_asynccnt 0x0" ::: "memory");
#endif
#endif
}

// ---------------------------------------------------------------------------
// f32 -> bf16 conversion (grid-stride)
// ---------------------------------------------------------------------------
__global__ __launch_bounds__(TPB)
void f32_to_bf16_k(const float* __restrict__ x, __bf16* __restrict__ y, long long n) {
  long long i = (long long)blockIdx.x * TPB + threadIdx.x;
  long long s = (long long)gridDim.x * TPB;
  for (; i < n; i += s) y[i] = (__bf16)x[i];
}

// ---------------------------------------------------------------------------
// Tiled bf16 WMMA GEMM:  C[M,N] (f32) = A[M,K] * op(B)
//   TRANSB=false : B is K x N row-major (NN)
//   TRANSB=true  : B is N x K row-major (NT, i.e. C = A * B^T)
// Block tile 64x64, K-step 64, 128 threads = 4 waves, each wave 32x32
// (2x2 accumulators, fragment reuse across 4 WMMAs). Double-buffered LDS,
// staged with async global->LDS DMA when available.
// Requires M%64==0, N%64==0, K%64==0 (true for every GEMM in this model).
// grid = (N/64, M/64, batch); strides in elements.
// ---------------------------------------------------------------------------
template <bool TRANSB>
__global__ __launch_bounds__(GEMM_TPB)
void gemm_bf16_wmma(const __bf16* __restrict__ A, long long sAb,
                    const __bf16* __restrict__ B, long long sBb,
                    float* __restrict__ C, long long sCb,
                    int M, int N, int K) {
  // 72 = 64 + pad(8): row stride 144B (16B-aligned, bank-conflict skewed)
  __shared__ __align__(16) __bf16 As[2][64][72];
  __shared__ __align__(16) __bf16 Bs[2][64][72];

  const int tid  = threadIdx.x;
  const int lane = tid & 31;
  const int wv   = tid >> 5;     // 0..3
  const int h    = lane >> 4;    // half-wave select (K sub-range per ISA layout)
  const int l16  = lane & 15;

  const int m0 = blockIdx.y * 64;
  const int n0 = blockIdx.x * 64;
  A += (long long)blockIdx.z * sAb + (long long)m0 * K;
  B += (long long)blockIdx.z * sBb + (TRANSB ? (long long)n0 * K : 0);
  C += (long long)blockIdx.z * sCb;

  const int warpM = (wv & 1) * 32;
  const int warpN = (wv >> 1) * 32;

  v8f acc[2][2] = {};

  // staging coords: each thread moves 64B of A (and of B when NT)
  const int srow = tid >> 1;        // 0..63
  const int scol = (tid & 1) * 32;  // 0 or 32

  auto stage = [&](int buf, int k0) {
    const __bf16* ga = A + (long long)srow * K + k0 + scol;
#pragma unroll
    for (int j = 0; j < 4; ++j)
      copy16_g2l(ga + j * 8, &As[buf][srow][scol + j * 8]);
    if (TRANSB) {
      const __bf16* gb = B + (long long)srow * K + k0 + scol;
#pragma unroll
      for (int j = 0; j < 4; ++j)
        copy16_g2l(gb + j * 8, &Bs[buf][srow][scol + j * 8]);
    } else {
      // B is K x N row-major: coalesced vec8 read along N, transpose into LDS
#pragma unroll
      for (int r = 0; r < 4; ++r) {
        const int c  = tid + GEMM_TPB * r;  // 0..511
        const int kk = c >> 3;              // 0..63
        const int nn = (c & 7) * 8;         // 0..56
        union { uint4v v; __bf16 e[8]; } u;
        u.v = *(const uint4v*)(B + (long long)(k0 + kk) * N + n0 + nn);
#pragma unroll
        for (int j = 0; j < 8; ++j) Bs[buf][nn + j][kk] = u.e[j];
      }
    }
  };

  auto compute = [&](int buf) {
#pragma unroll
    for (int ks = 0; ks < 64; ks += 32) {
      // Fragment layouts per CDNA5 ISA 7.12.2:
      // A 16x32 bf16: lane(l16)=row; lanes0-15: K 0-7 & 16-23; lanes16-31: K 8-15 & 24-31
      // B 32x16 bf16: lane(l16)=col; lanes0-15: K 0-7 & 16-23; lanes16-31: K 8-15 & 24-31
      v16bf a0, a1, b0, b1;
      union { uint4v u[2]; v16bf f; } t;
      t.u[0] = *(const uint4v*)(&As[buf][warpM + l16][ks + h * 8]);
      t.u[1] = *(const uint4v*)(&As[buf][warpM + l16][ks + h * 8 + 16]);
      a0 = t.f;
      t.u[0] = *(const uint4v*)(&As[buf][warpM + 16 + l16][ks + h * 8]);
      t.u[1] = *(const uint4v*)(&As[buf][warpM + 16 + l16][ks + h * 8 + 16]);
      a1 = t.f;
      t.u[0] = *(const uint4v*)(&Bs[buf][warpN + l16][ks + h * 8]);
      t.u[1] = *(const uint4v*)(&Bs[buf][warpN + l16][ks + h * 8 + 16]);
      b0 = t.f;
      t.u[0] = *(const uint4v*)(&Bs[buf][warpN + 16 + l16][ks + h * 8]);
      t.u[1] = *(const uint4v*)(&Bs[buf][warpN + 16 + l16][ks + h * 8 + 16]);
      b1 = t.f;

      acc[0][0] = __builtin_amdgcn_wmma_f32_16x16x32_bf16(false, a0, false, b0,
                                                          (short)0, acc[0][0], false, false);
      acc[0][1] = __builtin_amdgcn_wmma_f32_16x16x32_bf16(false, a0, false, b1,
                                                          (short)0, acc[0][1], false, false);
      acc[1][0] = __builtin_amdgcn_wmma_f32_16x16x32_bf16(false, a1, false, b0,
                                                          (short)0, acc[1][0], false, false);
      acc[1][1] = __builtin_amdgcn_wmma_f32_16x16x32_bf16(false, a1, false, b1,
                                                          (short)0, acc[1][1], false, false);
    }
  };

  int buf = 0;
  stage(0, 0);
  wait_async_lds();
  __syncthreads();
  for (int k0 = 0; k0 < K; k0 += 64) {
    if (k0 + 64 < K) stage(buf ^ 1, k0 + 64);  // overlap DMA with WMMA
    compute(buf);
    wait_async_lds();
    __syncthreads();
    buf ^= 1;
  }

  // C 16x16 f32 layout: VGPR r -> row r + 8*h, col = l16
#pragma unroll
  for (int i = 0; i < 2; ++i) {
    const int row0 = m0 + warpM + i * 16 + 8 * h;
#pragma unroll
    for (int j = 0; j < 2; ++j) {
      const int col = n0 + warpN + j * 16 + l16;
#pragma unroll
      for (int r = 0; r < 8; ++r)
        C[(long long)(row0 + r) * N + col] = acc[i][j][r];
    }
  }
}

// ---------------------------------------------------------------------------
// Depthwise / 2-grouped 3x3 conv, SAME padding, 32x32 maps.
// in_per_out==1: out[o] = conv(x[o], w[o,0])
// in_per_out==2: g=o/2,  out[o] = conv(x[2g],w[o,0]) + conv(x[2g+1],w[o,1])
// grid = B*C blocks, 256 threads.
// ---------------------------------------------------------------------------
__global__ __launch_bounds__(TPB)
void dwconv3x3_k(const float* __restrict__ x, const float* __restrict__ w,
                 float* __restrict__ y, int C, int in_per_out) {
  const int idx = blockIdx.x;
  const int b = idx / C;
  const int o = idx % C;
  const int icBase = (in_per_out == 2) ? (o & ~1) : o;

  __shared__ float tile[2][32][33];
  for (int ic = 0; ic < in_per_out; ++ic) {
    const float* src = x + (((long long)b * C + icBase + ic) << 10);
    for (int i = threadIdx.x; i < 1024; i += TPB)
      tile[ic][i >> 5][i & 31] = src[i];
  }
  __syncthreads();

  float wv[2][9];
  for (int ic = 0; ic < in_per_out; ++ic)
#pragma unroll
    for (int j = 0; j < 9; ++j)
      wv[ic][j] = w[((long long)o * in_per_out + ic) * 9 + j];

  float* dst = y + (((long long)b * C + o) << 10);
  for (int i = threadIdx.x; i < 1024; i += TPB) {
    const int py = i >> 5, px = i & 31;
    float acc = 0.f;
    for (int ic = 0; ic < in_per_out; ++ic) {
#pragma unroll
      for (int ky = 0; ky < 3; ++ky) {
        const int yy = py + ky - 1;
        if (yy < 0 || yy > 31) continue;
#pragma unroll
        for (int kx = 0; kx < 3; ++kx) {
          const int xx = px + kx - 1;
          if (xx < 0 || xx > 31) continue;
          acc = fmaf(tile[ic][yy][xx], wv[ic][ky * 3 + kx], acc);
        }
      }
    }
    dst[i] = acc;
  }
}

// ---------------------------------------------------------------------------
// Row L2-normalize (1024 elems per row), output bf16. grid = #rows.
// ---------------------------------------------------------------------------
__global__ __launch_bounds__(TPB)
void l2norm_bf16_k(const float* __restrict__ x, __bf16* __restrict__ y) {
  const long long base = (long long)blockIdx.x << 10;
  const int tid = threadIdx.x;
  __shared__ float red[TPB];
  float s = 0.f;
  for (int i = tid; i < 1024; i += TPB) { float v = x[base + i]; s += v * v; }
  red[tid] = s;
  __syncthreads();
  for (int o = TPB / 2; o > 0; o >>= 1) {
    if (tid < o) red[tid] += red[tid + o];
    __syncthreads();
  }
  const float inv = 1.0f / fmaxf(sqrtf(red[0]), 1e-12f);
  for (int i = tid; i < 1024; i += TPB) y[base + i] = (__bf16)(x[base + i] * inv);
}

// ---------------------------------------------------------------------------
// Instance-norm statistics over one batch slab of n f32 elements (x*scale).
// grid = B. stats[2b]=mean, stats[2b+1]=rsqrt(var+eps).
// ---------------------------------------------------------------------------
__global__ __launch_bounds__(TPB)
void instnorm_stats_k(const float* __restrict__ x, long long sb, long long n,
                      float scale, float* __restrict__ stats) {
  const int tid = threadIdx.x;
  x += (long long)blockIdx.x * sb;
  float s = 0.f, s2 = 0.f;
  for (long long i = tid; i < n; i += TPB) {
    const float v = x[i] * scale;
    s += v;
    s2 += v * v;
  }
  __shared__ float r1[TPB], r2[TPB];
  r1[tid] = s; r2[tid] = s2;
  __syncthreads();
  for (int o = TPB / 2; o > 0; o >>= 1) {
    if (tid < o) { r1[tid] += r1[tid + o]; r2[tid] += r2[tid + o]; }
    __syncthreads();
  }
  if (tid == 0) {
    const float m = r1[0] / (float)n;
    const float var = r2[0] / (float)n - m * m;
    stats[2 * blockIdx.x]     = m;
    stats[2 * blockIdx.x + 1] = rsqrtf(var + 1e-5f);
  }
}

// ---------------------------------------------------------------------------
// Fused instnorm-apply + row softmax, output bf16. Row length = 960.
// grid = B * rowsPerBatch.
// ---------------------------------------------------------------------------
__global__ __launch_bounds__(TPB)
void instnorm_softmax_k(const float* __restrict__ x, long long sb,
                        int rowsPerBatch, int ncols, float scale,
                        const float* __restrict__ stats,
                        __bf16* __restrict__ y, long long syb) {
  const int tid = threadIdx.x;
  const int b = blockIdx.x / rowsPerBatch;
  const int r = blockIdx.x % rowsPerBatch;
  const float* row = x + (long long)b * sb + (long long)r * ncols;
  __bf16* yrow = y + (long long)b * syb + (long long)r * ncols;
  const float m = stats[2 * b];
  const float rstd = stats[2 * b + 1];

  __shared__ float buf[960];
  __shared__ float red[TPB];

  float lmax = -3.4e38f;
  for (int i = tid; i < ncols; i += TPB) {
    const float v = (row[i] * scale - m) * rstd;
    buf[i] = v;
    lmax = fmaxf(lmax, v);
  }
  red[tid] = lmax;
  __syncthreads();
  for (int o = TPB / 2; o > 0; o >>= 1) {
    if (tid < o) red[tid] = fmaxf(red[tid], red[tid + o]);
    __syncthreads();
  }
  const float rmax = red[0];
  __syncthreads();

  float lsum = 0.f;
  for (int i = tid; i < ncols; i += TPB) {
    const float e = __expf(buf[i] - rmax);
    buf[i] = e;
    lsum += e;
  }
  red[tid] = lsum;
  __syncthreads();
  for (int o = TPB / 2; o > 0; o >>= 1) {
    if (tid < o) red[tid] += red[tid + o];
    __syncthreads();
  }
  const float inv = 1.0f / red[0];
  __syncthreads();
  for (int i = tid; i < ncols; i += TPB) yrow[i] = (__bf16)(buf[i] * inv);
}

// ---------------------------------------------------------------------------
// Host-side orchestration
// ---------------------------------------------------------------------------
static inline void conv_f32_bf16(const float* x, __bf16* y, long long n,
                                 hipStream_t st) {
  long long blocks = (n + (TPB * 8) - 1) / (TPB * 8);
  if (blocks > 4096) blocks = 4096;
  if (blocks < 1) blocks = 1;
  f32_to_bf16_k<<<dim3((unsigned)blocks), dim3(TPB), 0, st>>>(x, y, n);
}

extern "C" void kernel_launch(void* const* d_in, const int* in_sizes, int n_in,
                              void* d_out, int out_size, void* d_ws, size_t ws_size,
                              hipStream_t stream) {
  (void)in_sizes; (void)n_in; (void)out_size; (void)ws_size;

  const int B = 8, HW = 1024, KV = 960;
  const int Cs[4] = {64, 128, 256, 512};
  const float scale = 1.0f / sqrtf((float)KV);

  const float* emb[4]  = {(const float*)d_in[0], (const float*)d_in[1],
                          (const float*)d_in[2], (const float*)d_in[3]};
  const float* emb_all = (const float*)d_in[4];
  const float* w_mh[4] = {(const float*)d_in[5], (const float*)d_in[6],
                          (const float*)d_in[7], (const float*)d_in[8]};
  const float* w_q  = (const float*)d_in[9];
  const float* w_k  = (const float*)d_in[10];
  const float* w_v  = (const float*)d_in[11];
  const float* w_dq[4] = {(const float*)d_in[12], (const float*)d_in[13],
                          (const float*)d_in[14], (const float*)d_in[15]};
  const float* w_qc = (const float*)d_in[16];
  const float* w_kc = (const float*)d_in[17];
  const float* w_vc = (const float*)d_in[18];
  const float* w_pj[4] = {(const float*)d_in[19], (const float*)d_in[20],
                          (const float*)d_in[21], (const float*)d_in[22]};
  float* out = (float*)d_out;

  // ---- bump allocator over d_ws ----
  char* ws = (char*)d_ws;
  size_t off = 0;
  auto alloc = [&](size_t bytes) -> char* {
    char* p = ws + off;
    off = (off + bytes + 255) & ~(size_t)255;
    return p;
  };

  __bf16* wq_bf = (__bf16*)alloc((size_t)KV * KV * 2);
  __bf16* wk_bf = (__bf16*)alloc((size_t)KV * KV * 2);
  __bf16* wv_bf = (__bf16*)alloc((size_t)KV * KV * 2);
  __bf16* wmh_bf[4], *wpj_bf[4];
  for (int i = 0; i < 4; ++i) wmh_bf[i] = (__bf16*)alloc((size_t)Cs[i] * Cs[i] * 2);
  for (int i = 0; i < 4; ++i) wpj_bf[i] = (__bf16*)alloc((size_t)Cs[i] * Cs[i] * 2);
  __bf16* emb_all_bf = (__bf16*)alloc((size_t)B * KV * HW * 2);
  __bf16* emb_bf     = (__bf16*)alloc((size_t)B * 512 * HW * 2);  // reused as o_bf
  float*  tmp1       = (float*) alloc((size_t)B * KV * HW * 4);   // also ctx f32
  float*  tmp2       = (float*) alloc((size_t)B * KV * HW * 4);   // also sim/attn f32
  __bf16* q_bf       = (__bf16*)alloc((size_t)B * KV * HW * 2);
  __bf16* k_bf       = (__bf16*)alloc((size_t)B * KV * HW * 2);   // also p_bf
  __bf16* v_bf       = (__bf16*)alloc((size_t)B * KV * HW * 2);
  __bf16* ctx_bf     = (__bf16*)alloc((size_t)B * KV * HW * 2);
  __bf16* sims_bf    = (__bf16*)alloc((size_t)B * KV * KV * 2);
  float*  stats      = (float*) alloc((size_t)B * 2 * 4);

  // ---- convert weights + emb_all to bf16 ----
  conv_f32_bf16(w_q, wq_bf, (long long)KV * KV, stream);
  conv_f32_bf16(w_k, wk_bf, (long long)KV * KV, stream);
  conv_f32_bf16(w_v, wv_bf, (long long)KV * KV, stream);
  for (int i = 0; i < 4; ++i) {
    conv_f32_bf16(w_mh[i], wmh_bf[i], (long long)Cs[i] * Cs[i], stream);
    conv_f32_bf16(w_pj[i], wpj_bf[i], (long long)Cs[i] * Cs[i], stream);
  }
  conv_f32_bf16(emb_all, emb_all_bf, (long long)B * KV * HW, stream);

  const dim3 gblk(GEMM_TPB);
  const dim3 blk(TPB);
  const long long sX = (long long)KV * HW;   // per-batch activation stride

  // ---- Qc / Kc / Vc: conv1x1 -> dwconv3x3 -> (l2norm | convert) ----
  struct { const __bf16* w; const float* dw; __bf16* dst; bool norm; } qkv[3] = {
      {wq_bf, w_qc, q_bf, true},
      {wk_bf, w_kc, k_bf, true},
      {wv_bf, w_vc, v_bf, false}};
  for (int p = 0; p < 3; ++p) {
    gemm_bf16_wmma<false><<<dim3(HW / 64, KV / 64, B), gblk, 0, stream>>>(
        qkv[p].w, 0, emb_all_bf, sX, tmp1, sX, KV, HW, KV);
    dwconv3x3_k<<<dim3(B * KV), blk, 0, stream>>>(tmp1, qkv[p].dw, tmp2, KV, 1);
    if (qkv[p].norm)
      l2norm_bf16_k<<<dim3(B * KV), blk, 0, stream>>>(tmp2, qkv[p].dst);
    else
      conv_f32_bf16(tmp2, qkv[p].dst, (long long)B * KV * HW, stream);
  }

  // ---- sim = Qc * Kc^T  (NT), then instnorm+softmax -> bf16 ----
  const long long sSim = (long long)KV * KV;
  gemm_bf16_wmma<true><<<dim3(KV / 64, KV / 64, B), gblk, 0, stream>>>(
      q_bf, sX, k_bf, sX, tmp2, sSim, KV, KV, HW);
  instnorm_stats_k<<<dim3(B), blk, 0, stream>>>(tmp2, sSim, sSim, scale, stats);
  instnorm_softmax_k<<<dim3(B * KV), blk, 0, stream>>>(tmp2, sSim, KV, KV, scale,
                                                       stats, sims_bf, sSim);

  // ---- ctx = sim * Vc  (NN), convert to bf16 ----
  gemm_bf16_wmma<false><<<dim3(HW / 64, KV / 64, B), gblk, 0, stream>>>(
      sims_bf, sSim, v_bf, sX, tmp1, sX, KV, HW, KV);
  conv_f32_bf16(tmp1, ctx_bf, (long long)B * KV * HW, stream);

  // ---- branches ----
  long long outOff = 0;
  for (int i = 0; i < 4; ++i) {
    const int C = Cs[i];
    const long long sQ = (long long)C * HW;
    const long long sA = (long long)C * KV;

    conv_f32_bf16(emb[i], emb_bf, (long long)B * C * HW, stream);

    // q = l2norm(dwconv(conv1x1(emb, w_mh), w_q, groups=C/2))
    gemm_bf16_wmma<false><<<dim3(HW / 64, C / 64, B), gblk, 0, stream>>>(
        wmh_bf[i], 0, emb_bf, sQ, tmp1, sQ, C, HW, C);
    dwconv3x3_k<<<dim3(B * C), blk, 0, stream>>>(tmp1, w_dq[i], tmp2, C, 2);
    l2norm_bf16_k<<<dim3(B * C), blk, 0, stream>>>(tmp2, q_bf);

    // attn = q * ctx^T  (NT, K=HW)
    gemm_bf16_wmma<true><<<dim3(KV / 64, C / 64, B), gblk, 0, stream>>>(
        q_bf, sQ, ctx_bf, sX, tmp2, sA, C, KV, HW);
    instnorm_stats_k<<<dim3(B), blk, 0, stream>>>(tmp2, sA, sA, scale, stats);
    instnorm_softmax_k<<<dim3(B * C), blk, 0, stream>>>(tmp2, sA, C, KV, scale,
                                                        stats, k_bf, sA);

    // o = p * ctx  (NN, K=KV) -> f32, then bf16 (reuse emb_bf)
    gemm_bf16_wmma<false><<<dim3(HW / 64, C / 64, B), gblk, 0, stream>>>(
        k_bf, sA, ctx_bf, sX, tmp1, sQ, C, HW, KV);
    conv_f32_bf16(tmp1, emb_bf, (long long)B * C * HW, stream);

    // O = conv1x1(o, w_proj) straight into d_out
    gemm_bf16_wmma<false><<<dim3(HW / 64, C / 64, B), gblk, 0, stream>>>(
        wpj_bf[i], 0, emb_bf, sQ, out + outOff, sQ, C, HW, C);

    outOff += (long long)B * C * HW;
  }
}
